// EGNN_output_h_45707041964562
// MI455X (gfx1250) — compile-verified, matrix operand order checked by reading, not run on hardware
//
#include <hip/hip_runtime.h>
#include <math.h>

#define BSz 32
#define Nn  96
#define HID 64

typedef __attribute__((ext_vector_type(16))) _Float16 v16h;
typedef __attribute__((ext_vector_type(8)))  _Float16 v8h;
typedef __attribute__((ext_vector_type(8)))  float    v8f;

// fast SiLU: v_exp_f32 + v_rcp_f32 (~1 ulp rcp; inputs already f16-quantized)
__device__ __forceinline__ float silu_f(float v) {
    return v * __builtin_amdgcn_rcpf(1.0f + __expf(-v));
}

// K index inside a 16-bit A/B fragment: element e (0..15), lane-group g (0/1).
__device__ __forceinline__ int frag_k(int e, int g) {
    int v = e >> 1, p = e & 1;
    int k = (v < 4) ? (2 * v + p) : (16 + 2 * (v - 4) + p);
    return k + 8 * g;
}

// A fragment from row-major f16 (ld arbitrary): two contiguous 16B chunks.
__device__ __forceinline__ v16h load_a_rm(const _Float16* rowbase, int k0, int g) {
    const v8h c0 = *(const v8h*)(rowbase + k0 + 8 * g);
    const v8h c1 = *(const v8h*)(rowbase + k0 + 16 + 8 * g);
    v16h a;
#pragma unroll
    for (int e = 0; e < 8; ++e) { a[e] = c0[e]; a[8 + e] = c1[e]; }
    return a;
}

__device__ __forceinline__ v8f wmma16(v16h a, v16h b, v8f c) {
    return __builtin_amdgcn_wmma_f32_16x16x32_f16(false, a, false, b, (short)0, c,
                                                  false, false);
}

// ---------- weight pre-swizzle into B-fragment-contiguous layout ----------
// dst element ((kb*4+nt)*32 + lane)*16 + e  =  W[(kb*32+frag_k(e,g))*64 + nt*16+nlo]
struct PrepJobs {
    const float* src[20];
    int K[20];
    int dstOff[20];
};

__global__ void prep_weights_kernel(PrepJobs jobs, _Float16* __restrict__ dst) {
    const int j = blockIdx.y;
    const int elems = jobs.K[j] * 64;
    const float* __restrict__ src = jobs.src[j];
    _Float16* __restrict__ d = dst + jobs.dstOff[j];
    for (int idx = blockIdx.x * blockDim.x + threadIdx.x; idx < elems;
         idx += gridDim.x * blockDim.x) {
        int e = idx & 15, lane = (idx >> 4) & 31, frag = idx >> 9;
        int kb = frag >> 2, nt = frag & 3;
        int g = lane >> 4, nlo = lane & 15;
        int k = kb * 32 + frag_k(e, g);
        int n = nt * 16 + nlo;
        d[idx] = (_Float16)src[k * 64 + n];
    }
}

// ---------------- embedding: h0 = (h*nm) @ We + be ; x0 = x*nm ----------------
__global__ void emb_kernel(const float* __restrict__ x_in, const float* __restrict__ h_in,
                           const float* __restrict__ nm, const float* __restrict__ We,
                           const float* __restrict__ be, float* __restrict__ h0,
                           _Float16* __restrict__ h16, float* __restrict__ x0) {
    int t = blockIdx.x * blockDim.x + threadIdx.x;
    if (t >= BSz * Nn * HID) return;
    int node = t >> 6, f = t & 63;
    float m = nm[node];
    float acc = be[f];
#pragma unroll
    for (int k = 0; k < 16; ++k) acc += h_in[node * 16 + k] * m * We[k * HID + f];
    h0[t] = acc;
    h16[t] = (_Float16)acc;
    if (f < 3) x0[node * 3 + f] = x_in[node * 3 + f] * m;
}

// ---------------- output head ----------------
__global__ void emb_out_kernel(const float* __restrict__ h, const float* __restrict__ nm,
                               const float* __restrict__ Wo, const float* __restrict__ bo,
                               float* __restrict__ out) {
    int t = blockIdx.x * blockDim.x + threadIdx.x;
    if (t >= BSz * Nn * 16) return;
    int node = t >> 4, f = t & 15;
    float acc = bo[f];
#pragma unroll
    for (int k = 0; k < HID; ++k) acc += h[node * HID + k] * Wo[k * 16 + f];
    out[t] = acc * nm[node];
}

// ---------------- edge kernel: one workgroup per (b, i) row ----------------
__global__ void __launch_bounds__(192)
egnn_edge_kernel(const float* __restrict__ h_cur, const _Float16* __restrict__ h16_cur,
                 const float* __restrict__ x_cur, float* __restrict__ x_next,
                 float* __restrict__ agg_out,
                 const float* __restrict__ node_mask, const float* __restrict__ edge_mask,
                 const float* __restrict__ e1W, const float* __restrict__ e1b,
                 const float* __restrict__ e2b, const float* __restrict__ c1b,
                 const float* __restrict__ c2W,
                 const _Float16* __restrict__ whjS, const _Float16* __restrict__ w2S,
                 const _Float16* __restrict__ c1S) {
    __shared__ v16h sWhjV[256];             // 8 frags x 32 lanes (B-swizzled)
    __shared__ v16h sW2V[256];
    __shared__ v16h sWc1V[256];
    __shared__ _Float16 sHj[Nn * 64];       // h16[b, j, :]
    __shared__ _Float16 sEdgeBuf[Nn * 64];  // edge1-out then m (wave-private rows)
    __shared__ float sWrad[64], sRowc[64], sB2[64], sBc1[64], sC2[64];
    __shared__ float sHi[64];
    __shared__ float sRad[Nn], sDiff[Nn * 3], sEm[Nn];
    __shared__ float sAgg[64], sXacc[3];

    const int i = blockIdx.x, b = blockIdx.y;
    const int node = b * Nn + i;
    const int tid = threadIdx.x;

    // ---- phase 1: b128 staging of pre-swizzled weights + h16 row tile ----
    {
        uint4* d0 = (uint4*)sWhjV;
        uint4* d1 = (uint4*)sW2V;
        uint4* d2 = (uint4*)sWc1V;
        const uint4* s0 = (const uint4*)whjS;
        const uint4* s1 = (const uint4*)w2S;
        const uint4* s2 = (const uint4*)c1S;
        for (int idx = tid; idx < 512; idx += 192) {
            d0[idx] = s0[idx];
            d1[idx] = s1[idx];
            d2[idx] = s2[idx];
        }
        uint4* dh = (uint4*)sHj;
        const uint4* sh = (const uint4*)(h16_cur + (size_t)b * Nn * 64);
        for (int idx = tid; idx < 768; idx += 192) dh[idx] = sh[idx];
    }
    if (tid < 64) {
        sWrad[tid] = e1W[128 * 64 + tid];
        sB2[tid]   = e2b[tid];
        sBc1[tid]  = c1b[tid];
        sC2[tid]   = c2W[tid];
        sHi[tid]   = h_cur[node * 64 + tid];
        sAgg[tid]  = 0.0f;
    }
    if (tid < Nn) {
        float xi0 = x_cur[node * 3 + 0], xi1 = x_cur[node * 3 + 1], xi2 = x_cur[node * 3 + 2];
        const float* xj = x_cur + (b * Nn + tid) * 3;
        float d0 = xi0 - xj[0], d1 = xi1 - xj[1], d2 = xi2 - xj[2];
        float rad = d0 * d0 + d1 * d1 + d2 * d2;
        float inv = __builtin_amdgcn_rcpf(sqrtf(rad + 1e-8f) + 1.0f);
        sDiff[tid * 3 + 0] = d0 * inv;
        sDiff[tid * 3 + 1] = d1 * inv;
        sDiff[tid * 3 + 2] = d2 * inv;
        sRad[tid] = rad;
        sEm[tid]  = edge_mask[(size_t)node * Nn + tid];
    }
    if (tid < 3) sXacc[tid] = 0.0f;
    __syncthreads();

    // ---- phase 2: row-constant part of edge1 (h_i @ W_hi + b1) ----
    if (tid < 64) {
        float acc = e1b[tid];
        for (int k = 0; k < 64; ++k) acc += sHi[k] * e1W[k * 64 + tid];
        sRowc[tid] = acc;
    }
    __syncthreads();

    // ---- phase 3: WMMA edge pipeline ----
    {
        const int lane = tid & 31;
        const int wv   = tid >> 5;
        const int m0   = wv * 16;
        const int nlo  = lane & 15;
        const int g    = lane >> 4;
        const _Float16* rowH = sHj + (m0 + nlo) * 64;
        const _Float16* rowE = sEdgeBuf + (m0 + nlo) * 64;

        // edge1
        v16h a0 = load_a_rm(rowH, 0, g);
        v16h a1 = load_a_rm(rowH, 32, g);
#pragma unroll
        for (int nt = 0; nt < 4; ++nt) {
            const int n = nt * 16 + nlo;
            v8f acc = {};
            acc = wmma16(a0, sWhjV[nt * 32 + lane], acc);
            acc = wmma16(a1, sWhjV[(4 + nt) * 32 + lane], acc);
            float rc = sRowc[n], wr = sWrad[n];
#pragma unroll
            for (int r = 0; r < 8; ++r) {
                int ed = m0 + 8 * g + r;
                float v = acc[r] + rc + sRad[ed] * wr;
                sEdgeBuf[ed * 64 + n] = (_Float16)silu_f(v);
            }
        }
        asm volatile("s_wait_dscnt 0" ::: "memory");

        // edge2: m -> overwrite sEdgeBuf; column sums -> sAgg
        v16h s0 = load_a_rm(rowE, 0, g);
        v16h s1 = load_a_rm(rowE, 32, g);
#pragma unroll
        for (int nt = 0; nt < 4; ++nt) {
            const int n = nt * 16 + nlo;
            v8f acc = {};
            acc = wmma16(s0, sW2V[nt * 32 + lane], acc);
            acc = wmma16(s1, sW2V[(4 + nt) * 32 + lane], acc);
            float bb = sB2[n];
            float colsum = 0.0f;
#pragma unroll
            for (int r = 0; r < 8; ++r) {
                int ed = m0 + 8 * g + r;
                float v = silu_f(acc[r] + bb) * sEm[ed];
                colsum += v;
                sEdgeBuf[ed * 64 + n] = (_Float16)v;
            }
            atomicAdd(&sAgg[n], colsum);
        }
        asm volatile("s_wait_dscnt 0" ::: "memory");

        // coord1 + coord2 dot in registers; butterfly over 16 lanes
        v16h mA0 = load_a_rm(rowE, 0, g);
        v16h mA1 = load_a_rm(rowE, 32, g);
        float pw[8] = {0, 0, 0, 0, 0, 0, 0, 0};
#pragma unroll
        for (int nt = 0; nt < 4; ++nt) {
            const int n = nt * 16 + nlo;
            v8f acc = {};
            acc = wmma16(mA0, sWc1V[nt * 32 + lane], acc);
            acc = wmma16(mA1, sWc1V[(4 + nt) * 32 + lane], acc);
            float bb = sBc1[n], c2n = sC2[n];
#pragma unroll
            for (int r = 0; r < 8; ++r) pw[r] += silu_f(acc[r] + bb) * c2n;
        }
#pragma unroll
        for (int r = 0; r < 8; ++r) {
            float v = pw[r];
            v += __shfl_xor(v, 1, 32);
            v += __shfl_xor(v, 2, 32);
            v += __shfl_xor(v, 4, 32);
            v += __shfl_xor(v, 8, 32);
            pw[r] = v;
        }
        if (nlo == 0) {
#pragma unroll
            for (int r = 0; r < 8; ++r) {
                int ed = m0 + 8 * g + r;
                float wv2 = pw[r] * sEm[ed];
                atomicAdd(&sXacc[0], sDiff[ed * 3 + 0] * wv2);
                atomicAdd(&sXacc[1], sDiff[ed * 3 + 1] * wv2);
                atomicAdd(&sXacc[2], sDiff[ed * 3 + 2] * wv2);
            }
        }
    }
    __syncthreads();

    // ---- phase 4: export agg, update x ----
    if (tid < 64) agg_out[node * 64 + tid] = sAgg[tid];
    if (tid < 3) {
        float nmv = node_mask[node];
        x_next[node * 3 + tid] = (x_cur[node * 3 + tid] + sXacc[tid]) * nmv;
    }
}

// -------- node kernel: batched WMMA node MLP, 64 nodes / block, 4 waves --------
__global__ void __launch_bounds__(128)
egnn_node_kernel(const float* __restrict__ h_cur, const _Float16* __restrict__ h16_cur,
                 const float* __restrict__ agg,
                 float* __restrict__ h_next, _Float16* __restrict__ h16_next,
                 const float* __restrict__ nm,
                 const float* __restrict__ n1b, const float* __restrict__ n2b,
                 const _Float16* __restrict__ n1S, const _Float16* __restrict__ n2S) {
    __shared__ v16h sW1V[512];              // K=128: 16 frags
    __shared__ v16h sW2V[256];              // K=64:  8 frags
    __shared__ _Float16 sIn[64 * 128];      // cat(h, agg) f16
    __shared__ _Float16 sT1[64 * 64];       // silu(node1) f16
    __shared__ float sB1[64], sB2[64];

    const int base = blockIdx.x * 64;
    const int tid = threadIdx.x;
    {
        uint4* d1 = (uint4*)sW1V; const uint4* s1 = (const uint4*)n1S;
        for (int idx = tid; idx < 1024; idx += 128) d1[idx] = s1[idx];
        uint4* d2 = (uint4*)sW2V; const uint4* s2 = (const uint4*)n2S;
        for (int idx = tid; idx < 512; idx += 128) d2[idx] = s2[idx];
    }
    if (tid < 64) { sB1[tid] = n1b[tid]; sB2[tid] = n2b[tid]; }
    for (int idx = tid; idx < 64 * 64; idx += 128) {
        int nd = idx >> 6, k = idx & 63;
        sIn[nd * 128 + k]      = h16_cur[(base + nd) * 64 + k];
        sIn[nd * 128 + 64 + k] = (_Float16)agg[(base + nd) * 64 + k];
    }
    __syncthreads();

    const int lane = tid & 31, wv = tid >> 5;
    const int m0 = wv * 16, nlo = lane & 15, g = lane >> 4;
    const _Float16* rowIn = sIn + (m0 + nlo) * 128;
    v16h a0 = load_a_rm(rowIn, 0, g);
    v16h a1 = load_a_rm(rowIn, 32, g);
    v16h a2 = load_a_rm(rowIn, 64, g);
    v16h a3 = load_a_rm(rowIn, 96, g);
#pragma unroll
    for (int nt = 0; nt < 4; ++nt) {
        const int n = nt * 16 + nlo;
        v8f acc = {};
        acc = wmma16(a0, sW1V[(0 * 4 + nt) * 32 + lane], acc);
        acc = wmma16(a1, sW1V[(1 * 4 + nt) * 32 + lane], acc);
        acc = wmma16(a2, sW1V[(2 * 4 + nt) * 32 + lane], acc);
        acc = wmma16(a3, sW1V[(3 * 4 + nt) * 32 + lane], acc);
        float bb = sB1[n];
#pragma unroll
        for (int r = 0; r < 8; ++r) {
            int nd = m0 + 8 * g + r;
            sT1[nd * 64 + n] = (_Float16)silu_f(acc[r] + bb);
        }
    }
    asm volatile("s_wait_dscnt 0" ::: "memory");
    const _Float16* rowT = sT1 + (m0 + nlo) * 64;
    v16h t0 = load_a_rm(rowT, 0, g);
    v16h t1 = load_a_rm(rowT, 32, g);
#pragma unroll
    for (int nt = 0; nt < 4; ++nt) {
        const int n = nt * 16 + nlo;
        v8f acc = {};
        acc = wmma16(t0, sW2V[nt * 32 + lane], acc);
        acc = wmma16(t1, sW2V[(4 + nt) * 32 + lane], acc);
        float bb = sB2[n];
#pragma unroll
        for (int r = 0; r < 8; ++r) {
            int nd = base + m0 + 8 * g + r;
            float hv = (h_cur[nd * 64 + n] + acc[r] + bb) * nm[nd];
            h_next[nd * 64 + n] = hv;
            h16_next[nd * 64 + n] = (_Float16)hv;
        }
    }
}

// d_in: 0:x 1:h 2:node_mask 3:edge_mask 4:emb.W 5:emb.b 6:emb_out.W 7:emb_out.b
// then 11/layer: coord1.W coord1.b coord2.W edge1.W edge1.b edge2.W edge2.b
//                node1.W node1.b node2.W node2.b
extern "C" void kernel_launch(void* const* d_in, const int* in_sizes, int n_in,
                              void* d_out, int out_size, void* d_ws, size_t ws_size,
                              hipStream_t stream) {
    (void)in_sizes; (void)n_in; (void)out_size; (void)ws_size;
    const float* x_in = (const float*)d_in[0];
    const float* h_in = (const float*)d_in[1];
    const float* nm   = (const float*)d_in[2];
    const float* em   = (const float*)d_in[3];
    const float* embW = (const float*)d_in[4];
    const float* embB = (const float*)d_in[5];
    const float* outW = (const float*)d_in[6];
    const float* outB = (const float*)d_in[7];

    const int NH = BSz * Nn * HID;           // 196608
    float* h_a  = (float*)d_ws;
    float* h_b  = h_a + NH;
    float* x_a  = h_b + NH;                  // BSz*Nn*3
    float* x_b  = x_a + BSz * Nn * 3;
    float* agg  = x_b + BSz * Nn * 3;        // NH
    _Float16* h16_a = (_Float16*)(agg + NH);
    _Float16* h16_b = h16_a + NH;
    _Float16* wsw   = h16_b + NH;            // 24576 f16 per layer

    // ---- weight pre-swizzle (edge1.hj, edge2, coord1, node1, node2 per layer) ----
    PrepJobs jobs;
    for (int l = 0; l < 4; ++l) {
        void* const* P = d_in + 8 + l * 11;
        int j = l * 5;
        jobs.src[j + 0] = ((const float*)P[3]) + 64 * 64;  jobs.K[j + 0] = 64;   // edge1 hj
        jobs.src[j + 1] = (const float*)P[5];              jobs.K[j + 1] = 64;   // edge2
        jobs.src[j + 2] = (const float*)P[0];              jobs.K[j + 2] = 64;   // coord1
        jobs.src[j + 3] = (const float*)P[7];              jobs.K[j + 3] = 128;  // node1
        jobs.src[j + 4] = (const float*)P[9];              jobs.K[j + 4] = 64;   // node2
        int off = l * 24576;
        jobs.dstOff[j + 0] = off + 0;
        jobs.dstOff[j + 1] = off + 4096;
        jobs.dstOff[j + 2] = off + 8192;
        jobs.dstOff[j + 3] = off + 12288;
        jobs.dstOff[j + 4] = off + 20480;
    }
    prep_weights_kernel<<<dim3(32, 20), 256, 0, stream>>>(jobs, wsw);

    emb_kernel<<<(NH + 255) / 256, 256, 0, stream>>>(x_in, h_in, nm, embW, embB,
                                                     h_a, h16_a, x_a);

    const float* hc = h_a; float* hn = h_b;
    const _Float16* h16c = h16_a; _Float16* h16n = h16_b;
    const float* xc = x_a; float* xn = x_b;
    for (int l = 0; l < 4; ++l) {
        void* const* P = d_in + 8 + l * 11;
        const _Float16* lsw = wsw + l * 24576;
        egnn_edge_kernel<<<dim3(Nn, BSz), 192, 0, stream>>>(
            hc, h16c, xc, xn, agg, nm, em,
            (const float*)P[3], (const float*)P[4],    // edge1 W,b
            (const float*)P[6],                        // edge2 b
            (const float*)P[1],                        // coord1 b
            (const float*)P[2],                        // coord2 W
            lsw + 0, lsw + 4096, lsw + 8192);          // swizzled hj / e2 / c1
        egnn_node_kernel<<<BSz * Nn / 64, 128, 0, stream>>>(
            hc, h16c, agg, hn, h16n, nm,
            (const float*)P[8], (const float*)P[10],   // node1.b, node2.b
            lsw + 12288, lsw + 20480);                 // swizzled n1 / n2
        const float* th = hc; hc = hn; hn = (float*)th;
        const _Float16* t16 = h16c; h16c = h16n; h16n = (_Float16*)t16;
        const float* tx = xc; xc = xn; xn = (float*)tx;
    }
    emb_out_kernel<<<(BSz * Nn * 16 + 255) / 256, 256, 0, stream>>>(
        hc, nm, outW, outB, (float*)d_out);
}
